// LinearDeformableCrossAttn_62586263437615
// MI455X (gfx1250) — matrix operand order repaired and checked by reading, not compiled, not run on hardware
//
#include <hip/hip_runtime.h>
#include <hip/hip_bf16.h>

// ---- problem constants (B=8, C=256, H=W=64, nh=8, npts=4, dh=32) ----
#define BB 8
#define CC 256
#define HH 64
#define WW 64
#define NH 8
#define NP 4
#define DH 32
#define NN 4096              // H*W
#define NOFF 96              // nh*npts*3
#define BSTRIDE 264          // padded LDS row stride (halves): 528B = 132 dwords == 4 mod 64 banks

typedef _Float16 v16h __attribute__((ext_vector_type(16)));
typedef _Float16 v8h  __attribute__((ext_vector_type(8)));
typedef float    v8f  __attribute__((ext_vector_type(8)));

__device__ __forceinline__ v8f wmma_f16(v16h a, v16h b, v8f c) {
    // D = A(16x32 f16) * B(32x16 f16) + C(16x16 f32)
    return __builtin_amdgcn_wmma_f32_16x16x32_f16(
        /*neg_a=*/false, a, /*neg_b=*/false, b,
        /*c_mod=*/(short)0, c, /*reuse_a=*/false, /*reuse_b=*/false);
}

__device__ __forceinline__ v16h cat8(v8h lo, v8h hi) {
    return __builtin_shufflevector(lo, hi, 0,1,2,3,4,5,6,7,8,9,10,11,12,13,14,15);
}

// ---------------- prep: (B,C,N) f32 -> (B*N, C) f16 ----------------
__global__ __launch_bounds__(256) void k_transpose_q(const float* __restrict__ q,
                                                     _Float16* __restrict__ qT) {
    int idx = blockIdx.x * 256 + threadIdx.x;      // over B*N*C, c fastest (coalesced write)
    int c  = idx & (CC - 1);
    int bp = idx >> 8;                              // b*N + p
    int b  = bp >> 12;
    int p  = bp & (NN - 1);
    qT[idx] = (_Float16)q[(b * CC + c) * NN + p];
}

// weights: (K, N) f32 -> transposed (N, K) f16, K contiguous
__global__ __launch_bounds__(256) void k_cvt_f16_T(const float* __restrict__ src,
                                                   _Float16* __restrict__ dst,
                                                   int K, int N) {
    int idx = blockIdx.x * 256 + threadIdx.x;      // k fastest -> coalesced writes
    if (idx >= K * N) return;
    int k = idx % K;
    int n = idx / K;
    dst[n * K + k] = (_Float16)src[k * N + n];
}

// ------- kernel 1: GEMM1 (qf @ W_off + b_off) + softmax + sampling locs -------
// block = 64 threads (2 waves); each wave owns one 16-pixel M-tile.
__global__ __launch_bounds__(64) void k_offsets(const _Float16* __restrict__ qT,
                                                const _Float16* __restrict__ wOffT,
                                                const float* __restrict__ bOff,
                                                float* __restrict__ sp) {
    __shared__ __align__(16) _Float16 sWT[NOFF * BSTRIDE]; // 50688 B: W_off^T, padded rows
    __shared__ float sOw[2][16 * NOFF];                    // 12288 B: per-wave 16x96 result

    int tid  = threadIdx.x;
    int lane = tid & 31;
    int wv   = tid >> 5;

    // cooperative load of W_off^T: 96 rows x 256 halves (32 uint4 per row)
    for (int q = tid; q < NOFF * 32; q += 64) {
        int r = q >> 5, ch = q & 31;
        ((uint4*)(sWT + r * BSTRIDE))[ch] = ((const uint4*)(wOffT + r * CC))[ch];
    }
    __syncthreads();

    int tile  = blockIdx.x * 2 + wv;                 // 16-row tile of the 32768-row A
    int m16   = lane & 15;
    int kb    = (lane >> 4) << 3;                    // A K-offset: 0 or 8
    int khalf = (lane >> 4) << 4;                    // 0 or 16; unused (B also uses kb)
    const _Float16* Arow = qT + (tile * 16 + m16) * CC;

    v8f acc[6] = {};
    #pragma unroll
    for (int ks = 0; ks < 8; ++ks) {                 // K = 256 in steps of 32
        int k0 = ks * 32 + kb;
        v8h alo = *(const v8h*)(Arow + k0);
        v8h ahi = *(const v8h*)(Arow + k0 + 16);
        v16h a  = cat8(alo, ahi);
        #pragma unroll
        for (int t = 0; t < 6; ++t) {                // N = 96 in 6 tiles of 16
            const _Float16* bp = sWT + (t * 16 + m16) * BSTRIDE + ks * 32 + kb;
            v16h bf = cat8(*(const v8h*)bp, *(const v8h*)(bp + 16));  // 2x ds_load_b128
            acc[t] = wmma_f16(a, bf, acc[t]);
        }
    }

    // spill 16x96 result to LDS with bias for cross-lane softmax
    #pragma unroll
    for (int t = 0; t < 6; ++t) {
        int n = t * 16 + m16;
        float bias = bOff[n];
        #pragma unroll
        for (int r = 0; r < 8; ++r)
            sOw[wv][(kb + r) * NOFF + n] = acc[t][r] + bias;   // row m = kb + r
    }
    __syncthreads();

    // 128 (pixel, head) items per tile, 4 per lane
    #pragma unroll
    for (int i = 0; i < 4; ++i) {
        int item = i * 32 + lane;
        int m    = item >> 3;
        int head = item & 7;
        int P = tile * 16 + m;
        int b = P >> 12;
        int p = P & (NN - 1);
        int py = p >> 6, px = p & 63;
        float gx = -1.0f + 2.0f * (float)px / 63.0f;
        float gy = -1.0f + 2.0f * (float)py / 63.0f;

        float dxv[NP], dyv[NP], lw[NP];
        float mx = -1e30f;
        #pragma unroll
        for (int pt = 0; pt < NP; ++pt) {
            const float* r = &sOw[wv][m * NOFF + head * 12 + pt * 3];
            dxv[pt] = r[0]; dyv[pt] = r[1]; lw[pt] = r[2];
            mx = fmaxf(mx, lw[pt]);
        }
        float se = 0.0f;
        #pragma unroll
        for (int pt = 0; pt < NP; ++pt) { lw[pt] = __expf(lw[pt] - mx); se += lw[pt]; }
        float inv = 1.0f / se;

        float* o = sp + ((b * NH + head) * NN + p) * 12;
        #pragma unroll
        for (int pt = 0; pt < NP; ++pt) {
            float lx = gx + dxv[pt] * 0.1f;
            float ly = gy + dyv[pt] * 0.1f;
            o[pt * 3 + 0] = (lx + 1.0f) * 0.5f * 63.0f;      // xpix
            o[pt * 3 + 1] = (ly + 1.0f) * 0.5f * 63.0f;      // ypix
            o[pt * 3 + 2] = lw[pt] * inv;                    // weight
        }
    }
}

// ------- kernel 2: bilinear gather + weighted sum (lane = channel) -------
__device__ __forceinline__ float fetch_k(const float* __restrict__ kc, int y, int x) {
    bool v = (y >= 0) & (y < HH) & (x >= 0) & (x < WW);
    int yc = y < 0 ? 0 : (y > HH - 1 ? HH - 1 : y);
    int xc = x < 0 ? 0 : (x > WW - 1 ? WW - 1 : x);
    return v ? kc[yc * WW + xc] : 0.0f;
}

__global__ __launch_bounds__(256) void k_sample(const float* __restrict__ key,
                                                const float* __restrict__ sp,
                                                _Float16* __restrict__ of) {
    int wid  = blockIdx.x * 8 + (threadIdx.x >> 5);   // (b*nh)*N + p
    int lane = threadIdx.x & 31;                      // channel within head
    int bh = wid >> 12;
    int p  = wid & (NN - 1);
    const float* sb = sp + wid * 12;                  // wave-uniform params
    const float* kc = key + (bh * DH + lane) * NN;    // key_r[bh, c, :, :]

    float acc = 0.0f;
    #pragma unroll
    for (int pt = 0; pt < NP; ++pt) {
        float xp = sb[pt * 3 + 0];
        float yp = sb[pt * 3 + 1];
        float wv = sb[pt * 3 + 2];
        float x0f = floorf(xp), y0f = floorf(yp);
        int x0 = (int)x0f, y0 = (int)y0f;
        float wx = xp - x0f, wy = yp - y0f;
        float v00 = fetch_k(kc, y0,     x0);
        float v01 = fetch_k(kc, y0,     x0 + 1);
        float v10 = fetch_k(kc, y0 + 1, x0);
        float v11 = fetch_k(kc, y0 + 1, x0 + 1);
        acc += wv * ((1.0f - wy) * ((1.0f - wx) * v00 + wx * v01) +
                              wy * ((1.0f - wx) * v10 + wx * v11));
    }
    int b = bh >> 3, head = bh & 7;
    of[(b * NN + p) * CC + head * DH + lane] = (_Float16)acc;  // coalesced 64B/wave
}

// ------- kernel 3: GEMM2 out_feat @ W_proj + b_proj (64x64 tiles) -------
__global__ __launch_bounds__(128) void k_proj(const _Float16* __restrict__ of,
                                              const _Float16* __restrict__ wPT,
                                              const float* __restrict__ bP,
                                              float* __restrict__ out) {
    __shared__ __align__(16) _Float16 sBT[64 * BSTRIDE];   // 33792 B: W_proj^T panel, padded
    int tid = threadIdx.x;
    int n0  = blockIdx.y * 64;

    // cooperative panel load: 64 rows x 256 halves (32 uint4 per row) from wPT[n0..n0+63]
    #pragma unroll
    for (int i = 0; i < 16; ++i) {
        int q = tid + i * 128;
        int r = q >> 5, ch = q & 31;
        ((uint4*)(sBT + r * BSTRIDE))[ch] = ((const uint4*)(wPT + (n0 + r) * CC))[ch];
    }
    __syncthreads();

    int w = tid >> 5, lane = tid & 31;
    int m16   = lane & 15;
    int kb    = (lane >> 4) << 3;
    int khalf = (lane >> 4) << 4;
    int m0 = blockIdx.x * 64 + w * 16;
    const _Float16* Arow = of + (m0 + m16) * CC;

    v8f acc[4] = {};
    #pragma unroll
    for (int ks = 0; ks < 8; ++ks) {
        int k0 = ks * 32 + kb;
        if (ks < 7) __builtin_prefetch(Arow + k0 + 32, 0, 1);  // global_prefetch_b8
        v8h alo = *(const v8h*)(Arow + k0);
        v8h ahi = *(const v8h*)(Arow + k0 + 16);
        v16h a  = cat8(alo, ahi);
        #pragma unroll
        for (int t = 0; t < 4; ++t) {
            const _Float16* bp = sBT + (t * 16 + m16) * BSTRIDE + ks * 32 + kb;
            v16h bf = cat8(*(const v8h*)bp, *(const v8h*)(bp + 16));  // 2x ds_load_b128
            acc[t] = wmma_f16(a, bf, acc[t]);
        }
    }

    #pragma unroll
    for (int t = 0; t < 4; ++t) {
        int n = n0 + t * 16 + m16;
        float bias = bP[n];
        #pragma unroll
        for (int r = 0; r < 8; ++r)
            out[(m0 + kb + r) * CC + n] = acc[t][r] + bias;
    }
}

// --------------------------- host launcher ---------------------------
extern "C" void kernel_launch(void* const* d_in, const int* in_sizes, int n_in,
                              void* d_out, int out_size, void* d_ws, size_t ws_size,
                              hipStream_t stream) {
    const float* query  = (const float*)d_in[0];   // (B, C, H, W)
    const float* key    = (const float*)d_in[1];   // (B, C, H, W)
    const float* W_off  = (const float*)d_in[2];   // (C, 96)
    const float* b_off  = (const float*)d_in[3];   // (96,)
    const float* W_proj = (const float*)d_in[4];   // (C, C)
    const float* b_proj = (const float*)d_in[5];   // (C,)
    float* out = (float*)d_out;                    // (B, N, C) f32

    char* ws = (char*)d_ws;
    _Float16* qT    = (_Float16*)(ws);                                   // 16 MB
    _Float16* wOffT = (_Float16*)(ws + (size_t)(16 << 20));              // 48 KB (96x256, K-contig)
    _Float16* wPT   = (_Float16*)(ws + (size_t)(16 << 20) + (64 << 10)); // 128 KB (256x256, K-contig)
    float*    sp    = (float*)   (ws + (size_t)(16 << 20) + (256 << 10)); // 12.6 MB
    _Float16* of    = (_Float16*)(ws + (size_t)(30 << 20));              // 16 MB

    // prep: query transpose to (B*N, C) f16; weights to transposed f16
    k_transpose_q<<<(BB * NN * CC) / 256, 256, 0, stream>>>(query, qT);
    k_cvt_f16_T<<<(CC * NOFF + 255) / 256, 256, 0, stream>>>(W_off, wOffT, CC, NOFF);
    k_cvt_f16_T<<<(CC * CC + 255) / 256, 256, 0, stream>>>(W_proj, wPT, CC, CC);

    // GEMM1 + softmax + sampling locations: 2048 tiles, 2 per block
    k_offsets<<<1024, 64, 0, stream>>>(qT, wOffT, b_off, sp);

    // bilinear sampling: 262144 waves, 8 per block
    k_sample<<<(BB * NH * NN) / 8, 256, 0, stream>>>(key, sp, of);

    // GEMM2: M=32768, N=256, K=256
    dim3 g3((BB * NN) / 64, CC / 64);
    k_proj<<<g3, 128, 0, stream>>>(of, wPT, b_proj, out);
}
